// Model_64209761075494
// MI455X (gfx1250) — matrix operand order repaired and checked by reading, not compiled, not run on hardware
//
#include <hip/hip_runtime.h>
#include <math.h>

typedef __attribute__((ext_vector_type(16))) __bf16 v16bf;
typedef __attribute__((ext_vector_type(8)))  float  v8f;

#define B_      32
#define L1_     512
#define L2_     16
#define D_      300
#define H_      300
#define G_      900      // 3H
#define C_      3
#define KT_     10       // k-tiles: 300 padded to 320
#define NT_D    19       // n-tiles for N=300
#define NT_G    57       // n-tiles for N=900
#define LDC_D   304
#define LDC_G   912
#define ROWS_CTX (B_*L1_)   // 16384
#define ROWS_ASP (B_*L2_)   // 512
#define RT_CTX  (ROWS_CTX/16)
#define RT_ASP  (ROWS_ASP/16)

#define XSLICE  (16 * LDC_G)          // floats per prefetched xg time-slice
#define XCHUNKS (16 * LDC_G * 4 / 16) // 3648 b128 chunks per slice

// ---------------- embedding gather ----------------
__global__ void k_gather(const int* __restrict__ ids, const float* __restrict__ emb,
                         float* __restrict__ out, int rows) {
  int row = blockIdx.x;
  if (row >= rows) return;
  const float* src = emb + (size_t)ids[row] * D_;
  float* dst = out + (size_t)row * D_;
  for (int d = threadIdx.x; d < D_; d += blockDim.x) dst[d] = src[d];
}

// ---------------- pack A fragments (16x32 MxK bf16, ISA wave32 layout) ----------------
__global__ void k_pack_a(const float* __restrict__ src, __bf16* __restrict__ dst,
                         int rowTiles, int srcStride, int K) {
  int total = rowTiles * KT_ * 512;
  for (int idx = blockIdx.x * blockDim.x + threadIdx.x; idx < total;
       idx += gridDim.x * blockDim.x) {
    int li = idx & 511, tile = idx >> 9;
    int kt = tile % KT_, rt = tile / KT_;
    int lane = li >> 4, e = li & 15;
    int g = lane >> 4, m = rt * 16 + (lane & 15);
    int k = kt * 32 + ((e < 8) ? (e + 8 * g) : (16 + (e - 8) + 8 * g));
    float v = (k < K) ? src[(size_t)m * srcStride + k] : 0.f;
    dst[idx] = (__bf16)v;
  }
}

// ---------------- pack B fragments (32x16 KxN bf16): B[k][n] = W[n][k] ----------------
__global__ void k_pack_b(const float* __restrict__ W, __bf16* __restrict__ dst,
                         int nTiles, int N, int K) {
  int total = KT_ * nTiles * 512;
  for (int idx = blockIdx.x * blockDim.x + threadIdx.x; idx < total;
       idx += gridDim.x * blockDim.x) {
    int li = idx & 511, tile = idx >> 9;
    int nt = tile % nTiles, kt = tile / nTiles;
    int lane = li >> 4, e = li & 15;
    int g = lane >> 4, n = nt * 16 + (lane & 15);
    int k = kt * 32 + ((e < 8) ? (e + 8 * g) : (16 + (e - 8) + 8 * g));
    float v = (n < N && k < K) ? W[(size_t)n * K + k] : 0.f;
    dst[idx] = (__bf16)v;
  }
}

// ------- WMMA GEMM: one wave per (1 row-tile x 4 n-tiles); A reused 4x in regs -------
__global__ __launch_bounds__(256) void k_gemm(const __bf16* __restrict__ A,
    const __bf16* __restrict__ Bf, const float* __restrict__ bias,
    float* __restrict__ C, int rowTiles, int nTiles, int Nout, int ldc) {
  int ngrp = (nTiles + 3) >> 2;
  int wave = blockIdx.x * 8 + (threadIdx.x >> 5);
  if (wave >= rowTiles * ngrp) return;
  int rt = wave / ngrp;
  int n0 = (wave - rt * ngrp) * 4;
  int lane = threadIdx.x & 31;
  int n1 = (n0 + 1 < nTiles) ? n0 + 1 : nTiles - 1;
  int n2 = (n0 + 2 < nTiles) ? n0 + 2 : nTiles - 1;
  int n3 = (n0 + 3 < nTiles) ? n0 + 3 : nTiles - 1;
  v8f acc0 = {}, acc1 = {}, acc2 = {}, acc3 = {};
  const __bf16* ap = A + ((size_t)rt * KT_) * 512 + lane * 16;
  const __bf16* bp = Bf + (size_t)lane * 16;
#pragma unroll
  for (int kt = 0; kt < KT_; ++kt) {
    v16bf a = *(const v16bf*)(ap + (size_t)kt * 512);
    const __bf16* bk = bp + (size_t)kt * nTiles * 512;
    v16bf b0 = *(const v16bf*)(bk + (size_t)n0 * 512);
    v16bf b1 = *(const v16bf*)(bk + (size_t)n1 * 512);
    v16bf b2 = *(const v16bf*)(bk + (size_t)n2 * 512);
    v16bf b3 = *(const v16bf*)(bk + (size_t)n3 * 512);
    acc0 = __builtin_amdgcn_wmma_f32_16x16x32_bf16(false, a, false, b0, (short)0, acc0, false, false);
    acc1 = __builtin_amdgcn_wmma_f32_16x16x32_bf16(false, a, false, b1, (short)0, acc1, false, false);
    acc2 = __builtin_amdgcn_wmma_f32_16x16x32_bf16(false, a, false, b2, (short)0, acc2, false, false);
    acc3 = __builtin_amdgcn_wmma_f32_16x16x32_bf16(false, a, false, b3, (short)0, acc3, false, false);
  }
  int g = lane >> 4, col = lane & 15;
  auto st = [&](const v8f& acc, int nt) {
    int n = nt * 16 + col;
    if (n < Nout) {
      float bv = bias ? bias[n] : 0.f;
      for (int e = 0; e < 8; ++e)
        C[(size_t)(rt * 16 + e + 8 * g) * ldc + n] = acc[e] + bv;
    }
  };
  st(acc0, n0);
  if (n0 + 1 < nTiles) st(acc1, n0 + 1);
  if (n0 + 2 < nTiles) st(acc2, n0 + 2);
  if (n0 + 3 < nTiles) st(acc3, n0 + 3);
}

// ---------------- fused Bahdanau attention + beta logits (one wave per (b,l)) -------
__global__ __launch_bounds__(256) void k_attn(const float* __restrict__ cBuf,
    const float* __restrict__ aBuf, const float* __restrict__ aspF,
    const float* __restrict__ cwBuf, const float* __restrict__ Vatt,
    const float* __restrict__ Vw, const float* __restrict__ bVw,
    float* __restrict__ betaLogit) {
  int wave = blockIdx.x * 8 + (threadIdx.x >> 5);
  if (wave >= ROWS_CTX) return;
  int lane = threadIdx.x & 31;
  int b = wave >> 9;
  const float* crow  = cBuf  + (size_t)wave * LDC_D;
  const float* cwrow = cwBuf + (size_t)wave * LDC_D;
  float s[L2_];
  for (int m = 0; m < L2_; ++m) {
    const float* arow = aBuf + (size_t)(b * L2_ + m) * LDC_D;
    float p = 0.f;
    for (int d = lane; d < D_; d += 32) p += Vatt[d] * tanhf(crow[d] + arow[d]);
    for (int off = 16; off; off >>= 1) p += __shfl_xor(p, off, 32);
    s[m] = p;
  }
  float mx = s[0];
  for (int m = 1; m < L2_; ++m) mx = fmaxf(mx, s[m]);
  float sum = 0.f;
  for (int m = 0; m < L2_; ++m) { s[m] = expf(s[m] - mx); sum += s[m]; }
  float inv = 1.f / sum;
  float q = 0.f;
  for (int d = lane; d < D_; d += 32) {
    float cd = 0.f;
    for (int m = 0; m < L2_; ++m) cd += s[m] * aspF[(size_t)(b * L2_ + m) * D_ + d];
    float aa = tanhf(cwrow[d] + cd * inv);   // blin folded into cwBuf
    q += Vw[d] * aa;
  }
  for (int off = 16; off; off >>= 1) q += __shfl_xor(q, off, 32);
  if (lane == 0) betaLogit[wave] = q + bVw[0];
}

// ---------------- beta softmax over L1 per batch ----------------
__global__ __launch_bounds__(256) void k_beta(const float* __restrict__ logits,
                                              float* __restrict__ beta) {
  __shared__ float red[256];
  int b = blockIdx.x, tid = threadIdx.x;
  const float* x = logits + (size_t)b * L1_;
  float x0 = x[tid], x1 = x[tid + 256];
  red[tid] = fmaxf(x0, x1); __syncthreads();
  for (int s = 128; s; s >>= 1) { if (tid < s) red[tid] = fmaxf(red[tid], red[tid + s]); __syncthreads(); }
  float bmax = red[0]; __syncthreads();
  float e0 = expf(x0 - bmax), e1 = expf(x1 - bmax);
  red[tid] = e0 + e1; __syncthreads();
  for (int s = 128; s; s >>= 1) { if (tid < s) red[tid] += red[tid + s]; __syncthreads(); }
  float inv = 1.f / red[0];
  beta[(size_t)b * L1_ + tid] = e0 * inv;
  beta[(size_t)b * L1_ + tid + 256] = e1 * inv;
}

// ---------------- persistent GRU: 4 blocks = 2 dirs x 2 batch halves ----------------
// Dynamic LDS (175.6 KB < 320 KB/WGP):
//   hfrag  [KT_*512]    bf16   10240 B   A-fragments of h
//   hg     [16*LDC_G]   bf16   29184 B   h@whh^T staging
//   hcur   [16*LDC_D]   f32    19456 B   current hidden state
//   xbuf   [2][16*LDC_G] f32  116736 B   double-buffered async xg time-slices
__global__ __launch_bounds__(512) void k_gru(
    const float* __restrict__ xgF, const float* __restrict__ xgB,
    const __bf16* __restrict__ whhFf, const __bf16* __restrict__ whhFb,
    const float* __restrict__ bhhF, const float* __restrict__ bhhB,
    float* __restrict__ hsF, float* __restrict__ hsB) {
  int dir = blockIdx.x & 1;
  int b0 = (blockIdx.x >> 1) * 16;
  const float* xg  = dir ? xgB  : xgF;
  const __bf16* whh = dir ? whhFb : whhFf;
  const float* bhh = dir ? bhhB : bhhF;
  float* hs        = dir ? hsB  : hsF;

  extern __shared__ __align__(16) char smem[];
  __bf16* hfrag = (__bf16*)smem;                       // 10240 B
  __bf16* hg    = (__bf16*)(smem + 10240);             // 29184 B
  float*  hcur  = (float*)(smem + 10240 + 29184);      // 19456 B
  float*  xbuf  = (float*)(smem + 58880);              // 2 x 58368 B

  int tid = threadIdx.x;
  for (int i = tid; i < KT_ * 512;  i += 512) hfrag[i] = (__bf16)0.f;
  for (int i = tid; i < 16 * LDC_D; i += 512) hcur[i]  = 0.f;

  // async prefetch of xg time-slice t into LDS buffer (GLOBAL_LOAD_ASYNC_TO_LDS_B128)
  auto prefetch = [&](int t, int buf) {
    float* dstBuf = xbuf + buf * XSLICE;
    for (int idx = tid; idx < XCHUNKS; idx += 512) {
      int bi  = idx / 228;                 // 228 = LDC_G*4/16 chunks per row
      int off = (idx - bi * 228) * 16;     // byte offset within row
      const char* src = (const char*)(xg + ((size_t)(b0 + bi) * L1_ + t) * LDC_G) + off;
      unsigned dst = (unsigned)(size_t)(void*)dstBuf + (unsigned)(bi * LDC_G * 4 + off);
      asm volatile("global_load_async_to_lds_b128 %0, %1, off"
                   :: "v"(dst), "v"(src) : "memory");
    }
  };

  prefetch(dir ? L1_ - 1 : 0, 0);
  asm volatile("s_wait_asynccnt 0x0" ::: "memory");
  __syncthreads();

  int wave = tid >> 5, lane = tid & 31, g = lane >> 4, col = lane & 15;
  int n0 = wave * 4;                                   // 4 n-tiles per wave (64 >= 57)
  int m0 = (n0     < NT_G) ? n0     : NT_G - 1;
  int m1 = (n0 + 1 < NT_G) ? n0 + 1 : NT_G - 1;
  int m2 = (n0 + 2 < NT_G) ? n0 + 2 : NT_G - 1;
  int m3 = (n0 + 3 < NT_G) ? n0 + 3 : NT_G - 1;

  for (int step = 0; step < L1_; ++step) {
    int t = dir ? (L1_ - 1 - step) : step;
    int cur = step & 1, nxt = cur ^ 1;
    if (step + 1 < L1_) prefetch(dir ? (L1_ - 2 - step) : (step + 1), nxt);

    // hg = h @ whh^T : each wave computes 4 adjacent 16x16 tiles, A reused from LDS
    {
      v8f a0 = {}, a1 = {}, a2 = {}, a3 = {};
#pragma unroll
      for (int kt = 0; kt < KT_; ++kt) {
        v16bf a = *(const v16bf*)(hfrag + kt * 512 + lane * 16);
        const __bf16* bk = whh + (size_t)kt * NT_G * 512 + lane * 16;
        v16bf bb0 = *(const v16bf*)(bk + (size_t)m0 * 512);
        v16bf bb1 = *(const v16bf*)(bk + (size_t)m1 * 512);
        v16bf bb2 = *(const v16bf*)(bk + (size_t)m2 * 512);
        v16bf bb3 = *(const v16bf*)(bk + (size_t)m3 * 512);
        a0 = __builtin_amdgcn_wmma_f32_16x16x32_bf16(false, a, false, bb0, (short)0, a0, false, false);
        a1 = __builtin_amdgcn_wmma_f32_16x16x32_bf16(false, a, false, bb1, (short)0, a1, false, false);
        a2 = __builtin_amdgcn_wmma_f32_16x16x32_bf16(false, a, false, bb2, (short)0, a2, false, false);
        a3 = __builtin_amdgcn_wmma_f32_16x16x32_bf16(false, a, false, bb3, (short)0, a3, false, false);
      }
      auto st = [&](const v8f& acc, int nt) {
        if (nt < NT_G) {
          int n = nt * 16 + col;
          if (n < G_)
            for (int e = 0; e < 8; ++e) hg[(e + 8 * g) * LDC_G + n] = (__bf16)acc[e];
        }
      };
      st(a0, n0); st(a1, n0 + 1); st(a2, n0 + 2); st(a3, n0 + 3);
    }
    __syncthreads();

    // gate math (f32): xg slice comes from the async-staged LDS buffer
    const float* xb = xbuf + cur * XSLICE;
    for (int idx = tid; idx < 16 * H_; idx += 512) {
      int bi = idx / H_, hh = idx - bi * H_;
      float rx = xb[bi * LDC_G + hh];
      float zx = xb[bi * LDC_G + H_ + hh];
      float nx = xb[bi * LDC_G + 2 * H_ + hh];
      float rh = (float)hg[bi * LDC_G + hh]          + bhh[hh];
      float zh = (float)hg[bi * LDC_G + H_ + hh]     + bhh[H_ + hh];
      float nh = (float)hg[bi * LDC_G + 2 * H_ + hh] + bhh[2 * H_ + hh];
      float h  = hcur[bi * LDC_D + hh];
      float r  = 1.f / (1.f + expf(-(rx + rh)));
      float z  = 1.f / (1.f + expf(-(zx + zh)));
      float nn = tanhf(nx + r * nh);
      float hn = (1.f - z) * nn + z * h;
      hcur[bi * LDC_D + hh] = hn;
      hs[((size_t)(b0 + bi) * L1_ + t) * H_ + hh] = hn;
    }
    __syncthreads();

    // rebuild A-fragments of h for next step
    for (int idx = tid; idx < KT_ * 512; idx += 512) {
      int kt = idx >> 9, li = idx & 511;
      int ln = li >> 4, e = li & 15;
      int gg = ln >> 4, m = ln & 15;
      int k = kt * 32 + ((e < 8) ? (e + 8 * gg) : (16 + (e - 8) + 8 * gg));
      hfrag[idx] = (__bf16)((k < H_) ? hcur[m * LDC_D + k] : 0.f);
    }
    // ensure next timestep's async xg slice has landed before buffers flip
    asm volatile("s_wait_asynccnt 0x0" ::: "memory");
    __syncthreads();
  }
}

// ---------------- senti = beta-weighted sum + output projection ----------------
__global__ __launch_bounds__(256) void k_out(const float* __restrict__ beta,
    const float* __restrict__ hsF, const float* __restrict__ hsB,
    const float* __restrict__ outW, const float* __restrict__ outb,
    float* __restrict__ out) {
  __shared__ float sb[L1_];
  __shared__ float ss[2 * H_];
  int b = blockIdx.x, tid = threadIdx.x;
  for (int l = tid; l < L1_; l += 256) sb[l] = beta[(size_t)b * L1_ + l];
  __syncthreads();
  for (int j = tid; j < 2 * H_; j += 256) {
    const float* base = (j < H_) ? (hsF + (size_t)b * L1_ * H_ + j)
                                 : (hsB + (size_t)b * L1_ * H_ + (j - H_));
    float acc = 0.f;
    for (int l = 0; l < L1_; ++l) acc += sb[l] * base[(size_t)l * H_];
    ss[j] = acc;
  }
  __syncthreads();
  int wave = tid >> 5, lane = tid & 31;
  if (wave < C_) {
    float acc = 0.f;
    for (int j = lane; j < 2 * H_; j += 32) acc += ss[j] * outW[wave * 2 * H_ + j];
    for (int off = 16; off; off >>= 1) acc += __shfl_xor(acc, off, 32);
    if (lane == 0) out[b * C_ + wave] = acc + outb[wave];
  }
}

extern "C" void kernel_launch(void* const* d_in, const int* in_sizes, int n_in,
                              void* d_out, int out_size, void* d_ws, size_t ws_size,
                              hipStream_t stream) {
  (void)in_sizes; (void)n_in; (void)out_size; (void)ws_size;
  const int*   ctx_ids = (const int*)d_in[0];
  const int*   asp_ids = (const int*)d_in[1];
  const float* emb   = (const float*)d_in[4];
  const float* Wc    = (const float*)d_in[5];
  const float* Wa    = (const float*)d_in[6];
  const float* Vatt  = (const float*)d_in[7];
  const float* Wlin  = (const float*)d_in[8];
  const float* blin  = (const float*)d_in[9];
  const float* Vw    = (const float*)d_in[10];
  const float* bVw   = (const float*)d_in[11];
  const float* wih_f = (const float*)d_in[12];
  const float* whh_f = (const float*)d_in[13];
  const float* bih_f = (const float*)d_in[14];
  const float* bhh_f = (const float*)d_in[15];
  const float* wih_b = (const float*)d_in[16];
  const float* whh_b = (const float*)d_in[17];
  const float* bih_b = (const float*)d_in[18];
  const float* bhh_b = (const float*)d_in[19];
  const float* outW  = (const float*)d_in[20];
  const float* outb  = (const float*)d_in[21];
  float* out = (float*)d_out;

  char* p = (char*)d_ws;
  auto take = [&](size_t bytes) { char* r = p; p += bytes; return (void*)r; };
  float*  ctxF     = (float*) take((size_t)ROWS_CTX * D_ * 4);
  float*  aspF     = (float*) take((size_t)ROWS_ASP * D_ * 4);
  __bf16* ctxFrag  = (__bf16*)take((size_t)RT_CTX * KT_ * 512 * 2);
  __bf16* aspFrag  = (__bf16*)take((size_t)RT_ASP * KT_ * 512 * 2);
  __bf16* WcF      = (__bf16*)take((size_t)KT_ * NT_D * 512 * 2);
  __bf16* WaF      = (__bf16*)take((size_t)KT_ * NT_D * 512 * 2);
  __bf16* WlinF    = (__bf16*)take((size_t)KT_ * NT_D * 512 * 2);
  __bf16* wihfF    = (__bf16*)take((size_t)KT_ * NT_G * 512 * 2);
  __bf16* wihbF    = (__bf16*)take((size_t)KT_ * NT_G * 512 * 2);
  __bf16* whhfF    = (__bf16*)take((size_t)KT_ * NT_G * 512 * 2);
  __bf16* whhbF    = (__bf16*)take((size_t)KT_ * NT_G * 512 * 2);
  float*  cBuf     = (float*) take((size_t)ROWS_CTX * LDC_D * 4);
  float*  aBuf     = (float*) take((size_t)ROWS_ASP * LDC_D * 4);
  float*  cwBuf    = (float*) take((size_t)ROWS_CTX * LDC_D * 4);
  float*  xgF      = (float*) take((size_t)ROWS_CTX * LDC_G * 4);
  float*  xgB      = (float*) take((size_t)ROWS_CTX * LDC_G * 4);
  float*  betaLg   = (float*) take((size_t)ROWS_CTX * 4);
  float*  beta     = (float*) take((size_t)ROWS_CTX * 4);
  float*  hsF      = (float*) take((size_t)ROWS_CTX * H_ * 4);
  float*  hsB      = (float*) take((size_t)ROWS_CTX * H_ * 4);

  k_gather<<<ROWS_CTX, 128, 0, stream>>>(ctx_ids, emb, ctxF, ROWS_CTX);
  k_gather<<<ROWS_ASP, 128, 0, stream>>>(asp_ids, emb, aspF, ROWS_ASP);

  k_pack_a<<<2048, 256, 0, stream>>>(ctxF, ctxFrag, RT_CTX, D_, D_);
  k_pack_a<<<128,  256, 0, stream>>>(aspF, aspFrag, RT_ASP, D_, D_);
  k_pack_b<<<64,  256, 0, stream>>>(Wc,    WcF,   NT_D, D_, D_);
  k_pack_b<<<64,  256, 0, stream>>>(Wa,    WaF,   NT_D, D_, D_);
  k_pack_b<<<64,  256, 0, stream>>>(Wlin,  WlinF, NT_D, D_, D_);
  k_pack_b<<<160, 256, 0, stream>>>(wih_f, wihfF, NT_G, G_, D_);
  k_pack_b<<<160, 256, 0, stream>>>(wih_b, wihbF, NT_G, G_, D_);
  k_pack_b<<<160, 256, 0, stream>>>(whh_f, whhfF, NT_G, G_, H_);
  k_pack_b<<<160, 256, 0, stream>>>(whh_b, whhbF, NT_G, G_, H_);

  int grpD = (NT_D + 3) / 4, grpG = (NT_G + 3) / 4;
  k_gemm<<<(RT_CTX * grpD + 7) / 8, 256, 0, stream>>>(ctxFrag, WcF,   nullptr, cBuf,  RT_CTX, NT_D, D_, LDC_D);
  k_gemm<<<(RT_ASP * grpD + 7) / 8, 256, 0, stream>>>(aspFrag, WaF,   nullptr, aBuf,  RT_ASP, NT_D, D_, LDC_D);
  k_gemm<<<(RT_CTX * grpD + 7) / 8, 256, 0, stream>>>(ctxFrag, WlinF, blin,    cwBuf, RT_CTX, NT_D, D_, LDC_D);
  k_gemm<<<(RT_CTX * grpG + 7) / 8, 256, 0, stream>>>(ctxFrag, wihfF, bih_f,   xgF,   RT_CTX, NT_G, G_, LDC_G);
  k_gemm<<<(RT_CTX * grpG + 7) / 8, 256, 0, stream>>>(ctxFrag, wihbF, bih_b,   xgB,   RT_CTX, NT_G, G_, LDC_G);

  k_attn<<<ROWS_CTX / 8, 256, 0, stream>>>(cBuf, aBuf, aspF, cwBuf, Vatt, Vw, bVw, betaLg);
  k_beta<<<B_, 256, 0, stream>>>(betaLg, beta);

  size_t gruLds = 10240 + 29184 + 19456 + 2 * (size_t)XSLICE * 4;  // 175616 B
  k_gru<<<4, 512, gruLds, stream>>>(xgF, xgB, whhfF, whhbF, bhh_f, bhh_b, hsF, hsB);

  k_out<<<B_, 256, 0, stream>>>(beta, hsF, hsB, outW, outb, out);
}
